// VectorQuantizer_8718783611237
// MI455X (gfx1250) — matrix operand order repaired and checked by reading, not compiled, not run on hardware
//
#include <hip/hip_runtime.h>
#include <hip/hip_bf16.h>

typedef __attribute__((ext_vector_type(16))) __bf16 v16bf;
typedef __attribute__((ext_vector_type(8)))  __bf16 v8bf;
typedef __attribute__((ext_vector_type(4)))  __bf16 v4bf;
typedef __attribute__((ext_vector_type(8)))  float  v8f;

#define N_ROWS   131072   // B*H*W
#define N_CODES  1024
#define CDIM     64
#define ZQ_ELEMS 8388608  // B*C*H*W
#define LOSS_OFF 8388608
#define IDX_OFF  8388609
#define LDS_EMB_BYTES (N_CODES * CDIM * 2)           // 128 KB bf16 codebook
#define LDS_TOTAL     (LDS_EMB_BYTES + N_CODES * 4)  // + 4 KB f32 norms

// ---------------------------------------------------------------------------
// Kernel 1: distance argmin via v_wmma_f32_16x16x32_bf16.
// Each block = 8 waves; each wave owns 16 rows of z_flat and scans all 1024
// codes in 16-wide column tiles (K=64 reduction = 2 WMMAs of K=32).
// ---------------------------------------------------------------------------
__global__ __launch_bounds__(256)
void vq_argmin_kernel(const float* __restrict__ z_e,
                      const float* __restrict__ emb,
                      float* __restrict__ out) {
    extern __shared__ char smem[];
    __bf16* ldsEmb  = (__bf16*)smem;
    float*  ldsNorm = (float*)(smem + LDS_EMB_BYTES);

    const int tid = threadIdx.x;

    // Stage codebook into LDS as bf16 (vectorized: float4 -> 4x bf16).
    const float4* e4 = (const float4*)emb;
    for (int i = tid; i < (N_CODES * CDIM) / 4; i += 256) {
        float4 v = e4[i];
        v4bf h;
        h[0] = (__bf16)v.x; h[1] = (__bf16)v.y;
        h[2] = (__bf16)v.z; h[3] = (__bf16)v.w;
        *(v4bf*)&ldsEmb[i * 4] = h;
    }
    // f32 code norms (embedding is L2-resident; read once per block).
    for (int k = tid; k < N_CODES; k += 256) {
        const float* ep = emb + k * CDIM;
        float s = 0.f;
#pragma unroll 8
        for (int c = 0; c < CDIM; ++c) s = fmaf(ep[c], ep[c], s);
        ldsNorm[k] = s;
    }
    __syncthreads();

    const int wave = tid >> 5;
    const int lane = tid & 31;
    const int m0   = blockIdx.x * 128 + wave * 16;

    // ---- A fragments (16x32 bf16, ISA layout) for K-chunks 0 and 1 ----
    // z_flat[m][c] = z_e[b*262144 + c*4096 + (m & 4095)], b = m >> 12
    const int m = m0 + (lane & 15);
    const float* zp = z_e + (size_t)(m >> 12) * 262144 + (m & 4095);
    const int chalf = (lane < 16) ? 0 : 8;     // lane-half K interleave
    v16bf a0, a1;
#pragma unroll
    for (int j = 0; j < 8; ++j) {
        a0[j]     = (__bf16)zp[(size_t)(chalf + j)      * 4096];
        a0[j + 8] = (__bf16)zp[(size_t)(16 + chalf + j) * 4096];
        a1[j]     = (__bf16)zp[(size_t)(32 + chalf + j) * 4096];
        a1[j + 8] = (__bf16)zp[(size_t)(48 + chalf + j) * 4096];
    }

    float best[8];
    int   bidx[8];
#pragma unroll
    for (int r = 0; r < 8; ++r) { best[r] = 3.4e38f; bidx[r] = 0; }

    const int ncol_lane = lane & 15;
    const int koff      = (lane < 16) ? 0 : 8;    // B: lane-half K interleave (table order)

    for (int nt = 0; nt < N_CODES / 16; ++nt) {
        const int n = nt * 16 + ncol_lane;
        const __bf16* bp = ldsEmb + n * CDIM + koff;
        v8bf p0 = *(const v8bf*)(bp + 0);
        v8bf p1 = *(const v8bf*)(bp + 16);
        v8bf p2 = *(const v8bf*)(bp + 32);
        v8bf p3 = *(const v8bf*)(bp + 48);
        v16bf b0 = __builtin_shufflevector(p0, p1, 0,1,2,3,4,5,6,7,8,9,10,11,12,13,14,15);
        v16bf b1 = __builtin_shufflevector(p2, p3, 0,1,2,3,4,5,6,7,8,9,10,11,12,13,14,15);

        v8f acc = {};
        acc = __builtin_amdgcn_wmma_f32_16x16x32_bf16(false, a0, false, b0,
                                                      (short)0, acc, false, false);
        acc = __builtin_amdgcn_wmma_f32_16x16x32_bf16(false, a1, false, b1,
                                                      (short)0, acc, false, false);

        const float en = ldsNorm[n];
#pragma unroll
        for (int r = 0; r < 8; ++r) {
            float d = fmaf(-2.f, acc[r], en);   // ||e||^2 - 2 z.e
            bool take = d < best[r];            // strict < => first occurrence wins
            best[r] = take ? d : best[r];
            bidx[r] = take ? n : bidx[r];
        }
    }

    // Cross-lane argmin within each 16-lane half (wave32 shuffles).
#pragma unroll
    for (int r = 0; r < 8; ++r) {
#pragma unroll
        for (int off = 8; off >= 1; off >>= 1) {
            float ov = __shfl_xor(best[r], off, 32);
            int   oi = __shfl_xor(bidx[r], off, 32);
            if (ov < best[r] || (ov == best[r] && oi < bidx[r])) {
                best[r] = ov; bidx[r] = oi;
            }
        }
        if ((lane & 15) == 0) {
            int row = m0 + r + ((lane >= 16) ? 8 : 0);
            out[IDX_OFF + row] = (float)bidx[r];
        }
    }
}

// ---------------------------------------------------------------------------
// Kernel 2: gather z_q, write straight-through output, block partial sums of
// (z_q - z_e)^2 into d_ws (deterministic fixed-order tree reduction).
// ---------------------------------------------------------------------------
__global__ __launch_bounds__(256)
void vq_gather_kernel(const float* __restrict__ z_e,
                      const float* __restrict__ emb,
                      float* __restrict__ out,
                      float* __restrict__ partials) {
    const int tid = blockIdx.x * 256 + threadIdx.x;   // over (B,C,H,W) layout
    const int w = tid & 63;
    const int h = (tid >> 6) & 63;
    const int c = (tid >> 12) & 63;
    const int b = tid >> 18;
    const int n = (b << 12) | (h << 6) | w;           // z_flat row

    const int k = (int)out[IDX_OFF + n];              // exact small integers
    const float zq = emb[k * CDIM + c];
    const float ze = z_e[tid];
    out[tid] = ze + (zq - ze);                        // same rounding as reference
    const float diff = zq - ze;
    float v = diff * diff;

#pragma unroll
    for (int off = 16; off >= 1; off >>= 1) v += __shfl_xor(v, off, 32);

    __shared__ float wsum[8];
    if ((threadIdx.x & 31) == 0) wsum[threadIdx.x >> 5] = v;
    __syncthreads();
    if (threadIdx.x == 0) {
        float s = 0.f;
#pragma unroll
        for (int i = 0; i < 8; ++i) s += wsum[i];
        partials[blockIdx.x] = s;
    }
}

// ---------------------------------------------------------------------------
// Kernel 3: deterministic final reduction -> vq_loss = 1.25 * mean(diff^2)
// (codebook and commitment losses are numerically identical).
// ---------------------------------------------------------------------------
__global__ __launch_bounds__(256)
void vq_loss_kernel(const float* __restrict__ partials, float* __restrict__ out) {
    float s = 0.f;
    for (int i = threadIdx.x; i < ZQ_ELEMS / 256; i += 256) s += partials[i];
#pragma unroll
    for (int off = 16; off >= 1; off >>= 1) s += __shfl_xor(s, off, 32);
    __shared__ float wsum[8];
    if ((threadIdx.x & 31) == 0) wsum[threadIdx.x >> 5] = s;
    __syncthreads();
    if (threadIdx.x == 0) {
        float t = 0.f;
#pragma unroll
        for (int i = 0; i < 8; ++i) t += wsum[i];
        out[LOSS_OFF] = 1.25f * t * (1.0f / (float)ZQ_ELEMS);
    }
}

extern "C" void kernel_launch(void* const* d_in, const int* in_sizes, int n_in,
                              void* d_out, int out_size, void* d_ws, size_t ws_size,
                              hipStream_t stream) {
    const float* z_e = (const float*)d_in[0];   // (32,64,64,64) f32
    const float* emb = (const float*)d_in[1];   // (1024,64) f32
    float* out = (float*)d_out;                 // [z_q_st | vq_loss | indices]
    float* partials = (float*)d_ws;             // 32768 floats

    vq_argmin_kernel<<<N_ROWS / 128, 256, LDS_TOTAL, stream>>>(z_e, emb, out);
    vq_gather_kernel<<<ZQ_ELEMS / 256, 256, 0, stream>>>(z_e, emb, out, partials);
    vq_loss_kernel<<<1, 256, 0, stream>>>(partials, out);
}
